// LoRAGNNAdapter_2929167695979
// MI455X (gfx1250) — matrix operand order repaired and checked by reading, not compiled, not run on hardware
//
#include <hip/hip_runtime.h>
#include <cstdint>

// ---------------------------------------------------------------------------
// Types for CDNA5 WMMA
// ---------------------------------------------------------------------------
typedef _Float16 v16h __attribute__((ext_vector_type(16)));
typedef _Float16 h8   __attribute__((ext_vector_type(8)));
typedef float    v8f  __attribute__((ext_vector_type(8)));

#define H_DIM 128
#define R_DIM 4
#define LORA_SCALE 2.0f
#define LN_EPS 1e-5f
#define PE_LEN 2000
#define K_PAD 136   // 128 + 8 f16 pad -> 272B row stride, kills LDS bank conflicts

// gfx1250 async global->LDS copies for weight staging (ASYNCcnt tracked).
#define TRY_ASYNC_LDS 1

__device__ __forceinline__ void stage16(const _Float16* g, _Float16* l) {
#if TRY_ASYNC_LDS
    asm volatile("global_load_async_to_lds_b128 %0, %1, off"
                 :: "v"((unsigned)(uintptr_t)l),
                    "v"((unsigned long long)(uintptr_t)g)
                 : "memory");
#else
    *(h8*)l = *(const h8*)g;
#endif
}

__device__ __forceinline__ void stage_drain() {
#if TRY_ASYNC_LDS
    asm volatile("s_wait_asynccnt 0x0" ::: "memory");
#endif
}

// ---------------------------------------------------------------------------
// Small utility kernels
// ---------------------------------------------------------------------------
__global__ void zero_f32_kernel(float* __restrict__ p, long long n) {
    long long i = (long long)blockIdx.x * blockDim.x + threadIdx.x;
    long long stride = (long long)gridDim.x * blockDim.x;
    for (; i < n; i += stride) p[i] = 0.0f;
}

__global__ void deg_kernel(const int* __restrict__ src, const int* __restrict__ dst,
                           float* __restrict__ deg_out, float* __restrict__ deg_in, int E) {
    int e = blockIdx.x * blockDim.x + threadIdx.x;
    if (e >= E) return;
    atomicAdd(deg_out + src[e], 1.0f);
    atomicAdd(deg_in  + dst[e], 1.0f);
}

__global__ void norm_kernel(float* __restrict__ p, int n) {
    int i = blockIdx.x * blockDim.x + threadIdx.x;
    if (i >= n) return;
    float d = p[i];
    p[i] = (d > 0.0f) ? rsqrtf(fmaxf(d, 1.0f)) : 0.0f;
}

// W_eff[n,k] = W[n,k] + 2 * sum_r B[n,r] * A[r,k]   (stored f16, row-major [128][K])
__global__ void fuse_lora_kernel(const float* __restrict__ W, const float* __restrict__ A,
                                 const float* __restrict__ Bm, _Float16* __restrict__ out, int K) {
    int idx = blockIdx.x * blockDim.x + threadIdx.x;
    if (idx >= H_DIM * K) return;
    int n = idx / K, k = idx - n * K;
    float acc = W[idx];
#pragma unroll
    for (int r = 0; r < R_DIM; ++r)
        acc += LORA_SCALE * Bm[n * R_DIM + r] * A[r * K + k];
    out[idx] = (_Float16)acc;
}

// Wt[n,k] = W[k,n]  (for agg @ W : B[k,n] = W[k,n] -> store transposed f16)
__global__ void transpose_w_kernel(const float* __restrict__ W, _Float16* __restrict__ out) {
    int idx = blockIdx.x * blockDim.x + threadIdx.x;
    if (idx >= H_DIM * H_DIM) return;
    int n = idx >> 7, k = idx & 127;
    out[idx] = (_Float16)W[k * H_DIM + n];
}

// ---------------------------------------------------------------------------
// Edge gather / scatter-add: agg[dst] += h[src] * norm_out[src]
// One edge per 32 lanes; lane handles 4 columns (float4 gather, 4 atomics).
// h fits in the 192MB L2, so gathers/atomics stay on-chip; prefetch a
// lookahead row to cover L2 latency on the random gather.
// ---------------------------------------------------------------------------
__global__ void scatter_edges_kernel(const float* __restrict__ h, const float* __restrict__ norm_out,
                                     const int* __restrict__ src, const int* __restrict__ dst,
                                     float* __restrict__ agg, int E) {
    long long idx = (long long)blockIdx.x * blockDim.x + threadIdx.x;
    int e = (int)(idx >> 5);
    if (e >= E) return;
    int q = (int)(idx & 31);
    if (q == 0) {
        int e2 = e + 256;
        if (e2 < E) __builtin_prefetch(h + (size_t)src[e2] * H_DIM, 0, 0);
    }
    int s = src[e], d = dst[e];
    float ns = norm_out[s];
    const float4 v = *(const float4*)(h + (size_t)s * H_DIM + q * 4);
    float* ap = agg + (size_t)d * H_DIM + q * 4;
    atomicAdd(ap + 0, v.x * ns);
    atomicAdd(ap + 1, v.y * ns);
    atomicAdd(ap + 2, v.z * ns);
    atomicAdd(ap + 3, v.w * ns);
}

// ---------------------------------------------------------------------------
// Fused multi-side WMMA GEMM (NSIDES = 1 or 2, compile-time):
//   Out = (rs1 ⊙ X1[:, kx1:+128]) @ B1 [+ X2[:, kx2:+128] @ B2] + bias1 [+ bias2] + pe
// Bi[k,n] = Wti[n, kwi + k]  (f16 row-major [128][wtldi]).
// 256 threads = 8 waves; each wave register-blocks TWO 16-row A tiles so every
// B fragment (2x ds_load_b128) feeds two independent WMMA accumulation chains.
// Block owns 256 rows. Weight tiles staged in padded LDS via async b128 copies.
// ---------------------------------------------------------------------------
__device__ __forceinline__ void pack8(v16h& f, int base, float4 a, float4 b, float s) {
    f[base + 0] = (_Float16)(a.x * s); f[base + 1] = (_Float16)(a.y * s);
    f[base + 2] = (_Float16)(a.z * s); f[base + 3] = (_Float16)(a.w * s);
    f[base + 4] = (_Float16)(b.x * s); f[base + 5] = (_Float16)(b.y * s);
    f[base + 6] = (_Float16)(b.z * s); f[base + 7] = (_Float16)(b.w * s);
}

// 16-bit A 16x32 layout: lane(half) holds K = 8*half+{0..7} and 16+8*half+{0..7}
__device__ __forceinline__ void load_afrags(v16h* afrag, const float* X, int ldx, int kx,
                                            int rclamp, int half, float rs) {
#pragma unroll
    for (int kc = 0; kc < 4; ++kc) {
        const float* xp = X + (size_t)rclamp * ldx + kx + kc * 32 + half * 8;
        float4 a0 = *(const float4*)(xp + 0);
        float4 a1 = *(const float4*)(xp + 4);
        float4 a2 = *(const float4*)(xp + 16);
        float4 a3 = *(const float4*)(xp + 20);
        pack8(afrag[kc], 0, a0, a1, rs);
        pack8(afrag[kc], 8, a2, a3, rs);
    }
}

template <int NSIDES>
__global__ __launch_bounds__(256)
void gemm_wmma_kernel(const float* __restrict__ X1, int ldx1, int kx1,
                      const float* __restrict__ rs1,
                      const _Float16* __restrict__ Wt1, int wtld1, int kw1,
                      const float* __restrict__ bias1,
                      const float* __restrict__ X2, int ldx2, int kx2,
                      const _Float16* __restrict__ Wt2, int wtld2, int kw2,
                      const float* __restrict__ bias2,
                      const float* __restrict__ pe,
                      float* __restrict__ Out, int n_rows) {
    __shared__ __align__(16) _Float16 wlds[NSIDES][H_DIM * K_PAD];

    // Stage weight tiles into padded LDS via async global->LDS b128 copies.
    for (int i = threadIdx.x; i < H_DIM * 16; i += 256) {
        int n = i >> 4, c8 = i & 15;
        stage16(Wt1 + (size_t)n * wtld1 + kw1 + c8 * 8, &wlds[0][n * K_PAD + c8 * 8]);
        if (NSIDES == 2)
            stage16(Wt2 + (size_t)n * wtld2 + kw2 + c8 * 8, &wlds[1][n * K_PAD + c8 * 8]);
    }
    stage_drain();
    __syncthreads();

    const int wave = threadIdx.x >> 5;
    const int lane = threadIdx.x & 31;
    const int row0 = blockIdx.x * 256 + wave * 32;       // wave owns 32 rows (2 tiles)
    if (row0 >= n_rows) return;          // wave-uniform: EXEC stays all-ones below

    const int m    = lane & 15;
    const int half = lane >> 4;

    // A fragments: [tile][side][kchunk]
    v16h af[2][NSIDES][4];
    int rclamp[2];
#pragma unroll
    for (int t = 0; t < 2; ++t) {
        int r = row0 + t * 16 + m;
        rclamp[t] = (r < n_rows) ? r : (n_rows - 1);
        const float rs = rs1 ? rs1[rclamp[t]] : 1.0f;
        load_afrags(af[t][0], X1, ldx1, kx1, rclamp[t], half, rs);
        if (NSIDES == 2)
            load_afrags(af[t][1], X2, ldx2, kx2, rclamp[t], half, 1.0f);
    }

    const int ncol = lane & 15;
#pragma unroll
    for (int nt = 0; nt < 8; ++nt) {
        v8f c0 = {}, c1 = {};
#pragma unroll
        for (int s = 0; s < NSIDES; ++s) {
#pragma unroll
            for (int kc = 0; kc < 4; ++kc) {
                // Dense 16-bit B 32x16: lane(half) holds K = 8*half+{0..7} and 16+8*half+{0..7}
                const _Float16* bp = &wlds[s][(nt * 16 + ncol) * K_PAD + kc * 32 + half * 8];
                h8 b0 = *(const h8*)(bp + 0);
                h8 b1 = *(const h8*)(bp + 16);
                v16h bfrag = __builtin_shufflevector(b0, b1, 0, 1, 2, 3, 4, 5, 6, 7,
                                                     8, 9, 10, 11, 12, 13, 14, 15);
                c0 = __builtin_amdgcn_wmma_f32_16x16x32_f16(
                        false, af[0][s][kc], false, bfrag, (short)0, c0, false, false);
                c1 = __builtin_amdgcn_wmma_f32_16x16x32_f16(
                        false, af[1][s][kc], false, bfrag, (short)0, c1, false, false);
            }
        }
        // Epilogue: C/D layout lane(ncol), VGPR v -> row m = v + 8*half
        const int col = nt * 16 + ncol;
#pragma unroll
        for (int t = 0; t < 2; ++t) {
            const v8f& c = t ? c1 : c0;
#pragma unroll
            for (int v = 0; v < 8; ++v) {
                int row = row0 + t * 16 + v + 8 * half;
                if (row < n_rows) {
                    float val = c[v];
                    if (bias1) val += bias1[col];
                    if (NSIDES == 2 && bias2) val += bias2[col];
                    if (pe) {
                        int p = row < PE_LEN ? row : (PE_LEN - 1);
                        val += pe[(size_t)p * H_DIM + col];
                    }
                    Out[(size_t)row * H_DIM + col] = val;
                }
            }
        }
    }
}

// ---------------------------------------------------------------------------
// h[row] += gelu(layernorm(t[row]) * gamma + beta)   — one wave per row
// ---------------------------------------------------------------------------
__global__ __launch_bounds__(256)
void ln_gelu_res_kernel(const float* __restrict__ t, const float* __restrict__ gamma,
                        const float* __restrict__ beta, float* __restrict__ h, int n) {
    int wave = threadIdx.x >> 5, lane = threadIdx.x & 31;
    int row = blockIdx.x * 8 + wave;
    if (row >= n) return;
    const float4 x = *(const float4*)(t + (size_t)row * H_DIM + lane * 4);
    float s = x.x + x.y + x.z + x.w;
#pragma unroll
    for (int o = 16; o > 0; o >>= 1) s += __shfl_xor(s, o, 32);
    float mu = s * (1.0f / 128.0f);
    float d0 = x.x - mu, d1 = x.y - mu, d2 = x.z - mu, d3 = x.w - mu;
    float vs = d0 * d0 + d1 * d1 + d2 * d2 + d3 * d3;
#pragma unroll
    for (int o = 16; o > 0; o >>= 1) vs += __shfl_xor(vs, o, 32);
    float rstd = rsqrtf(vs * (1.0f / 128.0f) + LN_EPS);
    int c0 = lane * 4;
    float4 g4 = *(const float4*)(gamma + c0);
    float4 b4 = *(const float4*)(beta + c0);
    float y0 = d0 * rstd * g4.x + b4.x;
    float y1 = d1 * rstd * g4.y + b4.y;
    float y2 = d2 * rstd * g4.z + b4.z;
    float y3 = d3 * rstd * g4.w + b4.w;
    const float k = 0.70710678118654752f;
    float4 hv = *(float4*)(h + (size_t)row * H_DIM + c0);
    hv.x += 0.5f * y0 * (1.0f + erff(y0 * k));
    hv.y += 0.5f * y1 * (1.0f + erff(y1 * k));
    hv.z += 0.5f * y2 * (1.0f + erff(y2 * k));
    hv.w += 0.5f * y3 * (1.0f + erff(y3 * k));
    *(float4*)(h + (size_t)row * H_DIM + c0) = hv;
}

// ---------------------------------------------------------------------------
// Host-side orchestration
// ---------------------------------------------------------------------------
extern "C" void kernel_launch(void* const* d_in, const int* in_sizes, int n_in,
                              void* d_out, int out_size, void* d_ws, size_t ws_size,
                              hipStream_t stream) {
    const float* features = (const float*)d_in[0];   // [N,256]
    const int*   edge_src = (const int*)d_in[1];     // [E]
    const int*   edge_dst = (const int*)d_in[2];     // [E]
    const float* pe       = (const float*)d_in[3];   // [2000,128]
    const float* W_in     = (const float*)d_in[4];   // [128,256]
    const float* b_in     = (const float*)d_in[5];
    const float* A_in     = (const float*)d_in[6];   // [4,256]
    const float* B_in     = (const float*)d_in[7];   // [128,4]
    const float* gnn_W    = (const float*)d_in[8];   // [2,128,128]
    const float* gnn_b    = (const float*)d_in[9];   // [2,128]
    const float* lW       = (const float*)d_in[10];  // [2,128,128]
    const float* lb       = (const float*)d_in[11];  // [2,128]
    const float* lA       = (const float*)d_in[12];  // [2,4,128]
    const float* lB       = (const float*)d_in[13];  // [2,128,4]
    const float* gamma    = (const float*)d_in[14];  // [2,128]
    const float* beta     = (const float*)d_in[15];  // [2,128]
    const float* W_out    = (const float*)d_in[16];  // [128,128]
    const float* b_out    = (const float*)d_in[17];
    const float* A_out    = (const float*)d_in[18];  // [4,128]
    const float* B_out    = (const float*)d_in[19];  // [128,4]

    const int N = in_sizes[0] / 256;
    const int E = in_sizes[1];
    const int IN_DIM = 256;

    // ---- workspace layout --------------------------------------------------
    auto align256 = [](size_t x) { return (x + 255) & ~(size_t)255; };
    char* ws = (char*)d_ws;
    size_t off = 0;
    float* norm_out = (float*)(ws + off); off += align256((size_t)N * 4);
    float* norm_in  = (float*)(ws + off); off += align256((size_t)N * 4);
    float* h        = (float*)(ws + off); off += align256((size_t)N * H_DIM * 4);
    float* agg      = (float*)(ws + off); off += align256((size_t)N * H_DIM * 4);
    _Float16* w_in_eff  = (_Float16*)(ws + off); off += align256((size_t)H_DIM * IN_DIM * 2);
    _Float16* w_gnn_t   = (_Float16*)(ws + off); off += align256((size_t)2 * H_DIM * H_DIM * 2);
    _Float16* w_l_eff   = (_Float16*)(ws + off); off += align256((size_t)2 * H_DIM * H_DIM * 2);
    _Float16* w_out_eff = (_Float16*)(ws + off); off += align256((size_t)H_DIM * H_DIM * 2);
    (void)ws_size; (void)n_in; (void)out_size;

    const int T = 256;
    const int gemm_blocks = (N + 255) / 256;

    // ---- degrees -> norms --------------------------------------------------
    zero_f32_kernel<<<1024, T, 0, stream>>>(norm_out, N);
    zero_f32_kernel<<<1024, T, 0, stream>>>(norm_in, N);
    deg_kernel<<<(E + T - 1) / T, T, 0, stream>>>(edge_src, edge_dst, norm_out, norm_in, E);
    norm_kernel<<<(N + T - 1) / T, T, 0, stream>>>(norm_out, N);
    norm_kernel<<<(N + T - 1) / T, T, 0, stream>>>(norm_in, N);

    // ---- fold LoRA into frozen weights, transpose gnn weights (f16) --------
    fuse_lora_kernel<<<(H_DIM * IN_DIM + T - 1) / T, T, 0, stream>>>(W_in, A_in, B_in, w_in_eff, IN_DIM);
    for (int i = 0; i < 2; ++i) {
        fuse_lora_kernel<<<(H_DIM * H_DIM + T - 1) / T, T, 0, stream>>>(
            lW + (size_t)i * H_DIM * H_DIM, lA + (size_t)i * R_DIM * H_DIM,
            lB + (size_t)i * H_DIM * R_DIM, w_l_eff + (size_t)i * H_DIM * H_DIM, H_DIM);
        transpose_w_kernel<<<(H_DIM * H_DIM + T - 1) / T, T, 0, stream>>>(
            gnn_W + (size_t)i * H_DIM * H_DIM, w_gnn_t + (size_t)i * H_DIM * H_DIM);
    }
    fuse_lora_kernel<<<(H_DIM * H_DIM + T - 1) / T, T, 0, stream>>>(W_out, A_out, B_out, w_out_eff, H_DIM);

    // ---- input projection: both K-halves in one fused pass + bias + PE -----
    gemm_wmma_kernel<2><<<gemm_blocks, T, 0, stream>>>(
        features, IN_DIM, 0,   nullptr, w_in_eff, IN_DIM, 0,   b_in,
        features, IN_DIM, 128,          w_in_eff, IN_DIM, 128, nullptr,
        pe, h, N);

    // ---- GNN layers --------------------------------------------------------
    long long scat_work = (long long)E * 32;
    int scat_blocks = (int)((scat_work + T - 1) / T);
    for (int i = 0; i < 2; ++i) {
        zero_f32_kernel<<<4096, T, 0, stream>>>(agg, (long long)N * H_DIM);
        scatter_edges_kernel<<<scat_blocks, T, 0, stream>>>(h, norm_out, edge_src, edge_dst, agg, E);
        // agg <- (agg * norm_in) @ gnn_W + gnn_b + h @ W_lora_eff + lb   (fused, in-place)
        gemm_wmma_kernel<2><<<gemm_blocks, T, 0, stream>>>(
            agg, H_DIM, 0, norm_in, w_gnn_t + (size_t)i * H_DIM * H_DIM, H_DIM, 0,
            gnn_b + (size_t)i * H_DIM,
            h,   H_DIM, 0,          w_l_eff + (size_t)i * H_DIM * H_DIM, H_DIM, 0,
            lb + (size_t)i * H_DIM,
            nullptr, agg, N);
        // h += gelu(layernorm(agg))
        ln_gelu_res_kernel<<<(N + 7) / 8, T, 0, stream>>>(
            agg, gamma + (size_t)i * H_DIM, beta + (size_t)i * H_DIM, h, N);
    }

    // ---- output projection (single side) -----------------------------------
    gemm_wmma_kernel<1><<<gemm_blocks, T, 0, stream>>>(
        h, H_DIM, 0, nullptr, w_out_eff, H_DIM, 0, b_out,
        nullptr, 0, 0, nullptr, 0, 0, nullptr,
        nullptr, (float*)d_out, N);
}
